// DualLayerSRFFN_90314572300882
// MI455X (gfx1250) — matrix operand order repaired and checked by reading, not compiled, not run on hardware
//
#include <hip/hip_runtime.h>
#include <hip/hip_bf16.h>
#include <stdint.h>

typedef __bf16 bf16_t;
typedef __attribute__((ext_vector_type(16))) __bf16 v16bf;
typedef __attribute__((ext_vector_type(8)))  float  v8f;

#define B_DIM 2048
#define D_DIM 4096
#define F_DIM 16384

#define T1_M 64
#define T1_N 64
#define T2_M 64
#define T2_N 128
#define BK   32
// 40 halves = 80 bytes row stride: 16B-aligned rows for b128 ops, and
// 20-word stride gives conflict-free 16-lane column walks (20*l mod 64 distinct).
#define LDSS 40

// Async global->LDS copy (CDNA5 VGLOBAL async op, tracked by ASYNCcnt).
// GV mode: vdst = LDS byte address (low 32 bits of the flat LDS pointer),
//          vaddr = 64-bit global address.
__device__ __forceinline__ void async_load_b128(const void* gptr, void* lptr) {
    const uint32_t lds_off = (uint32_t)(uintptr_t)lptr;       // flat LDS addr low bits = LDS offset
    const uint64_t gaddr   = (uint64_t)(uintptr_t)gptr;
    asm volatile("global_load_async_to_lds_b128 %0, %1, off"
                 :: "v"(lds_off), "v"(gaddr)
                 : "memory");
}
__device__ __forceinline__ void wait_asynccnt0() {
    asm volatile("s_wait_asynccnt 0" ::: "memory");
}

__device__ __forceinline__ float temporal_mix(float cur, float prev) {
    return (cur > prev) ? (cur + 0.1f * prev) : (prev + 0.1f * cur);
}
__device__ __forceinline__ float sign_act(float x) {   // THR == 0
    return (x > 0.f) ? 1.f : ((x < 0.f) ? -1.f : 0.f);
}

// ---------------------------------------------------------------------------
// Phase 1: 4 fused GEMM streams [B,D]x[D,F] + temporal mix + sign/gate
// writes fused (bf16, values in {-2..2}) to workspace
// ---------------------------------------------------------------------------
__global__ __launch_bounds__(256) void srffn_phase1(
    const float* __restrict__ sem, const float* __restrict__ phon,
    const float* __restrict__ prev_sa, const float* __restrict__ prev_pa,
    const float* __restrict__ prev_sr, const float* __restrict__ prev_pr,
    const float* __restrict__ w1, const float* __restrict__ wsr,
    const float* __restrict__ wpr, bf16_t* __restrict__ fused)
{
    __shared__ __align__(16) bf16_t lAs[2][T1_M][LDSS];   // sem rows   [m][k]
    __shared__ __align__(16) bf16_t lAp[2][T1_M][LDSS];   // phon rows  [m][k]
    __shared__ __align__(16) bf16_t lB1[2][T1_N][LDSS];   // w1  cols   [n][k] (transposed)
    __shared__ __align__(16) bf16_t lBs[2][T1_N][LDSS];   // wsr cols
    __shared__ __align__(16) bf16_t lBp[2][T1_N][LDSS];   // wpr cols

    const int tid   = threadIdx.x;
    const int lane  = tid & 31;
    const int wid   = tid >> 5;
    const int rowBlock = blockIdx.y * T1_M;
    const int colBlock = blockIdx.x * T1_N;
    const int waveRow  = (wid >> 2) * 32;   // 0 / 32
    const int waveCol  = (wid &  3) * 16;   // 0..48

    // staging geometry: A tiles 64x32 f32, B tiles 32x64 f32; 2 float4 each/thread
    const int ar  = tid >> 3;            // 0..31
    const int ac4 = (tid & 7) * 4;       // 0..28
    const int bk  = tid >> 4;            // 0..15
    const int bc4 = (tid & 15) * 4;      // 0..60

    float4 rAs[2], rAp[2], rB1[2], rBs[2], rBp[2];

    auto gload = [&](const float* base, long row, long col, long ld) -> float4 {
        return *reinterpret_cast<const float4*>(base + row * ld + col);
    };
    auto stage_load = [&](int k0) {
        #pragma unroll
        for (int h = 0; h < 2; ++h) {
            const int r = ar + 32 * h;
            rAs[h] = gload(sem,  rowBlock + r, k0 + ac4, D_DIM);
            rAp[h] = gload(phon, rowBlock + r, k0 + ac4, D_DIM);
            const int k = bk + 16 * h;
            rB1[h] = gload(w1,  k0 + k, colBlock + bc4, F_DIM);
            rBs[h] = gload(wsr, k0 + k, colBlock + bc4, F_DIM);
            rBp[h] = gload(wpr, k0 + k, colBlock + bc4, F_DIM);
        }
    };
    auto stage_store = [&](int buf) {
        #pragma unroll
        for (int h = 0; h < 2; ++h) {
            const int r = ar + 32 * h;
            const int k = bk + 16 * h;
            #pragma unroll
            for (int e = 0; e < 4; ++e) {
                lAs[buf][r][ac4 + e]   = (bf16_t)(&rAs[h].x)[e];
                lAp[buf][r][ac4 + e]   = (bf16_t)(&rAp[h].x)[e];
                lB1[buf][bc4 + e][k]   = (bf16_t)(&rB1[h].x)[e];
                lBs[buf][bc4 + e][k]   = (bf16_t)(&rBs[h].x)[e];
                lBp[buf][bc4 + e][k]   = (bf16_t)(&rBp[h].x)[e];
            }
        }
    };

    const int  lane16 = lane & 15;
    const bool hi     = (lane >= 16);
    const int  aKoff  = hi ? 16 : 0;    // A: lanes16-31 hold K {16..31}
    const int  bKoff  = hi ? 16 : 0;    // B: lanes16-31 hold K {16..31}

    auto load_a = [&](const bf16_t (*tile)[LDSS], int mrow) -> v16bf {
        const bf16_t* rp = &tile[mrow + lane16][0];
        v16bf a;
        #pragma unroll
        for (int j = 0; j < 16; ++j)
            a[j] = rp[j + aKoff];
        return a;
    };
    auto load_b = [&](const bf16_t (*tile)[LDSS], int ncol) -> v16bf {
        const bf16_t* rp = &tile[ncol + lane16][0];
        v16bf b;
        #pragma unroll
        for (int j = 0; j < 16; ++j)
            b[j] = rp[j + bKoff];
        return b;
    };

    v8f acc_sh[2] = {}, acc_ph[2] = {}, acc_sr[2] = {}, acc_pr[2] = {};

    stage_load(0);
    stage_store(0);
    const int nK = D_DIM / BK;
    for (int kk = 0; kk < nK; ++kk) {
        __syncthreads();
        const int buf = kk & 1;
        if (kk + 1 < nK) {
            stage_load((kk + 1) * BK);
            if (kk + 2 < nK) {  // L2 prefetch two steps ahead
                __builtin_prefetch(sem  + (long)(rowBlock + ar) * D_DIM + (kk + 2) * BK + ac4, 0, 1);
                __builtin_prefetch(w1   + (long)((kk + 2) * BK + bk) * F_DIM + colBlock + bc4, 0, 1);
            }
        }
        const v16bf b1 = load_b(lB1[buf], waveCol);
        const v16bf bs = load_b(lBs[buf], waveCol);
        const v16bf bp = load_b(lBp[buf], waveCol);
        #pragma unroll
        for (int t = 0; t < 2; ++t) {
            const v16bf as = load_a(lAs[buf], waveRow + 16 * t);
            const v16bf ap = load_a(lAp[buf], waveRow + 16 * t);
            acc_sh[t] = __builtin_amdgcn_wmma_f32_16x16x32_bf16(false, as, false, b1, (short)0, acc_sh[t], false, false);
            acc_ph[t] = __builtin_amdgcn_wmma_f32_16x16x32_bf16(false, ap, false, b1, (short)0, acc_ph[t], false, false);
            acc_sr[t] = __builtin_amdgcn_wmma_f32_16x16x32_bf16(false, as, false, bs, (short)0, acc_sr[t], false, false);
            acc_pr[t] = __builtin_amdgcn_wmma_f32_16x16x32_bf16(false, ap, false, bp, (short)0, acc_pr[t], false, false);
        }
        if (kk + 1 < nK) stage_store((kk + 1) & 1);
    }

    // epilogue: temporal mix + sign + gate -> fused (bf16)
    const long gRow0 = rowBlock + waveRow;
    const long gCol  = colBlock + waveCol + lane16;
    #pragma unroll
    for (int t = 0; t < 2; ++t) {
        #pragma unroll
        for (int i = 0; i < 8; ++i) {
            const long m   = gRow0 + 16 * t + i + (hi ? 8 : 0);
            const long idx = m * (long)F_DIM + gCol;
            const float msh = temporal_mix(acc_sh[t][i], prev_sa[idx]);
            const float mph = temporal_mix(acc_ph[t][i], prev_pa[idx]);
            const float msr = temporal_mix(acc_sr[t][i], prev_sr[idx]);
            const float mpr = temporal_mix(acc_pr[t][i], prev_pr[idx]);
            const float ss  = sign_act(msh);
            const float ps  = sign_act(mph);
            const float gs  = (msr > 0.f) ? ss : -ss;
            const float gp  = (mpr > 0.f) ? ps : -ps;
            fused[idx] = (bf16_t)(gs + gp);
        }
    }
}

// ---------------------------------------------------------------------------
// Phase 2: out = fused[B,F] @ w2[F,D]   (fused already bf16 in workspace)
// A tile staged with CDNA5 async global->LDS ops; B (w2) needs f32->bf16 so
// it round-trips through registers.
// ---------------------------------------------------------------------------
__global__ __launch_bounds__(256) void srffn_phase2(
    const bf16_t* __restrict__ fused, const float* __restrict__ w2,
    float* __restrict__ out)
{
    __shared__ __align__(16) bf16_t lA[2][T2_M][LDSS];   // fused rows [m][k]
    __shared__ __align__(16) bf16_t lB[2][T2_N][LDSS];   // w2 cols    [n][k]

    const int tid   = threadIdx.x;
    const int lane  = tid & 31;
    const int wid   = tid >> 5;
    const int rowBlock = blockIdx.y * T2_M;
    const int colBlock = blockIdx.x * T2_N;
    const int waveRow  = (wid >> 2) * 32;
    const int waveCol  = (wid &  3) * 32;

    // A tile: 64 rows x 32 bf16 = 256 x 16B chunks -> one async b128 per thread
    const int arr = tid >> 2;            // 0..63
    const int aq  = (tid & 3) * 8;       // half offset 0/8/16/24
    // B tile: 32 rows(k) x 128 cols f32 -> 4 float4 per thread
    float4 rB[4];

    auto stageA = [&](int buf, int k0) {
        const bf16_t* g = fused + (long)(rowBlock + arr) * F_DIM + k0 + aq;
        async_load_b128(g, &lA[buf][arr][aq]);
    };
    auto stageB_load = [&](int k0) {
        #pragma unroll
        for (int h = 0; h < 4; ++h) {
            const int idx = tid + 256 * h;
            const int k   = idx >> 5;            // 0..31
            const int q4  = (idx & 31) * 4;      // 0..124
            rB[h] = *reinterpret_cast<const float4*>(w2 + (long)(k0 + k) * D_DIM + colBlock + q4);
        }
    };
    auto stageB_store = [&](int buf) {
        #pragma unroll
        for (int h = 0; h < 4; ++h) {
            const int idx = tid + 256 * h;
            const int k   = idx >> 5;
            const int q4  = (idx & 31) * 4;
            #pragma unroll
            for (int e = 0; e < 4; ++e)
                lB[buf][q4 + e][k] = (bf16_t)(&rB[h].x)[e];
        }
    };

    const int  lane16 = lane & 15;
    const bool hi     = (lane >= 16);
    const int  aKoff  = hi ? 16 : 0;
    const int  bKoff  = hi ? 16 : 0;

    auto load_a = [&](const bf16_t (*tile)[LDSS], int mrow) -> v16bf {
        const bf16_t* rp = &tile[mrow + lane16][0];
        v16bf a;
        #pragma unroll
        for (int j = 0; j < 16; ++j)
            a[j] = rp[j + aKoff];
        return a;
    };
    auto load_b = [&](const bf16_t (*tile)[LDSS], int ncol) -> v16bf {
        const bf16_t* rp = &tile[ncol + lane16][0];
        v16bf b;
        #pragma unroll
        for (int j = 0; j < 16; ++j)
            b[j] = rp[j + bKoff];
        return b;
    };

    v8f acc[2][2] = {};

    stageA(0, 0);
    stageB_load(0);
    stageB_store(0);
    wait_asynccnt0();
    const int nK = F_DIM / BK;
    for (int kk = 0; kk < nK; ++kk) {
        __syncthreads();
        const int buf = kk & 1;
        if (kk + 1 < nK) {
            stageA(buf ^ 1, (kk + 1) * BK);
            stageB_load((kk + 1) * BK);
        }
        #pragma unroll
        for (int u = 0; u < 2; ++u) {
            const v16bf b = load_b(lB[buf], waveCol + 16 * u);
            #pragma unroll
            for (int t = 0; t < 2; ++t) {
                const v16bf a = load_a(lA[buf], waveRow + 16 * t);
                acc[t][u] = __builtin_amdgcn_wmma_f32_16x16x32_bf16(false, a, false, b, (short)0, acc[t][u], false, false);
            }
        }
        if (kk + 1 < nK) {
            stageB_store(buf ^ 1);
            wait_asynccnt0();   // async A-tile must land before next barrier publishes it
        }
    }

    const long gRow0 = rowBlock + waveRow;
    #pragma unroll
    for (int t = 0; t < 2; ++t) {
        #pragma unroll
        for (int u = 0; u < 2; ++u) {
            const long n = colBlock + waveCol + 16 * u + lane16;
            #pragma unroll
            for (int i = 0; i < 8; ++i) {
                const long m = gRow0 + 16 * t + i + (hi ? 8 : 0);
                out[m * (long)D_DIM + n] = acc[t][u][i];
            }
        }
    }
}

// ---------------------------------------------------------------------------
extern "C" void kernel_launch(void* const* d_in, const int* in_sizes, int n_in,
                              void* d_out, int out_size, void* d_ws, size_t ws_size,
                              hipStream_t stream) {
    (void)in_sizes; (void)n_in; (void)out_size; (void)ws_size;
    const float* sem      = (const float*)d_in[0];
    const float* phon     = (const float*)d_in[1];
    const float* prev_sa  = (const float*)d_in[2];
    const float* prev_pa  = (const float*)d_in[3];
    const float* prev_sr  = (const float*)d_in[4];
    const float* prev_pr  = (const float*)d_in[5];
    const float* w1       = (const float*)d_in[6];
    const float* w2       = (const float*)d_in[7];
    const float* wsr      = (const float*)d_in[8];
    const float* wpr      = (const float*)d_in[9];

    bf16_t* fused = (bf16_t*)d_ws;   // B*F bf16 = 64 MiB scratch
    float*  outp  = (float*)d_out;

    dim3 g1(F_DIM / T1_N, B_DIM / T1_M);   // (256, 32)
    srffn_phase1<<<g1, dim3(256), 0, stream>>>(sem, phon, prev_sa, prev_pa,
                                               prev_sr, prev_pr, w1, wsr, wpr, fused);
    dim3 g2(D_DIM / T2_N, B_DIM / T2_M);   // (32, 32)
    srffn_phase2<<<g2, dim3(256), 0, stream>>>(fused, w2, outp);
}
